// Attention_21122649162458
// MI455X (gfx1250) — matrix operand order repaired and checked by reading, not compiled, not run on hardware
//
#include <hip/hip_runtime.h>
#include <hip/hip_bf16.h>
#include <stdint.h>

typedef __attribute__((ext_vector_type(16))) __bf16 v16bf;
typedef __attribute__((ext_vector_type(8)))  float  v8f;

#define Bsz 32
#define Sdim 2048
#define Hdim 512

// ---------------------------------------------------------------------------
// Prep kernel 1: pack W_enc (rows 0..511 of W, fp32) into bf16 B-fragment
// layout.  WB[((nt*16 + kt)*32 + lane)*8 + v] packs W_enc[k][n], W_enc[k+1][n]
// with n = nt*16 + lane%16, k = kt*32 + (lane/16)*16 + 2v.
// ---------------------------------------------------------------------------
__global__ void prep_wb(const float* __restrict__ W, uint32_t* __restrict__ WB) {
    int idx  = blockIdx.x * 256 + threadIdx.x;      // 131072 threads
    int v    = idx & 7;
    int lane = (idx >> 3) & 31;
    int kt   = (idx >> 8) & 15;
    int nt   = idx >> 12;
    int half = lane >> 4;
    int n    = nt * 16 + (lane & 15);
    int k    = kt * 32 + half * 16 + 2 * v;
    __bf16 lo = (__bf16)W[(size_t)k * Hdim + n];
    __bf16 hi = (__bf16)W[(size_t)(k + 1) * Hdim + n];
    uint32_t plo = (uint32_t)__builtin_bit_cast(uint16_t, lo);
    uint32_t phi = (uint32_t)__builtin_bit_cast(uint16_t, hi);
    WB[idx] = plo | (phi << 16);
}

// ---------------------------------------------------------------------------
// Prep kernel 2: wcs[n] = sum_h W_cov[h][n]  (W rows 1024..1535)
// ---------------------------------------------------------------------------
__global__ void prep_wcs(const float* __restrict__ W, float* __restrict__ wcs) {
    int n = blockIdx.x * 256 + threadIdx.x;
    if (n >= Hdim) return;
    float s = 0.f;
    for (int h = 0; h < Hdim; ++h)
        s += W[(size_t)(1024 + h) * Hdim + n];
    wcs[n] = s;
}

// ---------------------------------------------------------------------------
// Prep kernel 3: dec_proj[b][k] = sum_e dec_input[b][e] * W_s[e][k]
// ---------------------------------------------------------------------------
__global__ void prep_decp(const float* __restrict__ W,
                          const float* __restrict__ dec,
                          float* __restrict__ decp) {
    int idx = blockIdx.x * 256 + threadIdx.x;       // 16384 threads
    int b = idx >> 9;
    int k = idx & 511;
    float s = 0.f;
    for (int e = 0; e < Hdim; ++e)
        s += dec[b * Hdim + e] * W[(size_t)(512 + e) * Hdim + k];
    decp[idx] = s;
}

// ---------------------------------------------------------------------------
// Async stage: copy the 16 KB B-fragment chunk for N-tile `nt` into LDS with
// global_load_async_to_lds_b128 (ASYNCcnt-tracked DMA).  256 threads, 64 B
// each (4 x b128).
// ---------------------------------------------------------------------------
__device__ __forceinline__ void stage_async(const uint32_t* WB, int nt,
                                            uint32_t lds_dst, int t) {
    uint64_t g = (uint64_t)(uintptr_t)(WB + (size_t)nt * 4096) + (uint32_t)t * 64u;
    uint32_t l = lds_dst + (uint32_t)t * 64u;
#pragma unroll
    for (int i = 0; i < 4; ++i) {
        uint32_t laddr = l + i * 16;
        uint64_t gaddr = g + i * 16;
        asm volatile("global_load_async_to_lds_b128 %0, %1, off"
                     :: "v"(laddr), "v"(gaddr) : "memory");
    }
}

// ---------------------------------------------------------------------------
// Main fused kernel: 8 waves/block, one 16-row M tile per wave.  A strip
// (16x512 bf16) in 128 VGPRs; B N-tiles staged to LDS once per block via
// async DMA (double buffered); v_wmma_f32_16x16x32_bf16 inner loop with a
// one-fragment-deep LDS software pipeline; fused tanh / dot(v_w) epilogue.
// ---------------------------------------------------------------------------
__global__ void __launch_bounds__(256) attn_main(
    const float*    __restrict__ enc,
    const uint32_t* __restrict__ WB,
    const float*    __restrict__ wcs,
    const float*    __restrict__ decp,
    const float*    __restrict__ cov,
    const float*    __restrict__ bias,
    const float*    __restrict__ vw,
    const float*    __restrict__ vb,
    float*          __restrict__ logits) {
    __shared__ uint32_t sB[2][4096];                // 2 x 16 KB double buffer

    const int t    = threadIdx.x;
    const int wave = t >> 5;
    const int lane = t & 31;
    const int half = lane >> 4;
    const int ln   = lane & 15;
    const int tile = blockIdx.x * 8 + wave;         // 0..4095
    const int row0 = tile * 16;                     // flattened (b,s) row
    const int b    = row0 >> 11;                    // / S

    const uint32_t lds_base = (uint32_t)(uintptr_t)&sB[0][0];

    // kick off DMA of the first B N-tile while we load/convert A
    stage_async(WB, 0, lds_base, t);

    // ---- A fragments: 16 k-tiles x v16bf (128 VGPRs); enc read exactly once
    const float* arow = enc + (size_t)(row0 + ln) * Hdim;
    v16bf a[16];
#pragma unroll
    for (int kt = 0; kt < 16; ++kt) {
        const int c1 = kt * 32 + half * 16;
        const int c2 = c1 + 8;
        float4 f0 = *(const float4*)(arow + c1);
        float4 f1 = *(const float4*)(arow + c1 + 4);
        float4 f2 = *(const float4*)(arow + c2);
        float4 f3 = *(const float4*)(arow + c2 + 4);
        v16bf tt;
        tt[0]  = (__bf16)f0.x; tt[1]  = (__bf16)f0.y;
        tt[2]  = (__bf16)f0.z; tt[3]  = (__bf16)f0.w;
        tt[4]  = (__bf16)f1.x; tt[5]  = (__bf16)f1.y;
        tt[6]  = (__bf16)f1.z; tt[7]  = (__bf16)f1.w;
        tt[8]  = (__bf16)f2.x; tt[9]  = (__bf16)f2.y;
        tt[10] = (__bf16)f2.z; tt[11] = (__bf16)f2.w;
        tt[12] = (__bf16)f3.x; tt[13] = (__bf16)f3.y;
        tt[14] = (__bf16)f3.z; tt[15] = (__bf16)f3.w;
        a[kt] = tt;
    }

    // coverage values for this lane's 8 output rows (m = j + 8*half)
    float covm[8];
#pragma unroll
    for (int j = 0; j < 8; ++j)
        covm[j] = cov[row0 + j + 8 * half];

    float att[8] = {0.f, 0.f, 0.f, 0.f, 0.f, 0.f, 0.f, 0.f};

    for (int nt = 0; nt < 32; ++nt) {
        // wait for this tile's DMA, make it visible to all 8 waves; the same
        // barrier guarantees the other buffer is no longer being read.
        asm volatile("s_wait_asynccnt 0" ::: "memory");
        __syncthreads();
        if (nt + 1 < 32)
            stage_async(WB, nt + 1, lds_base + ((nt + 1) & 1) * 16384u, t);

        const uint32_t* sbuf = &sB[nt & 1][0];
        const int n  = nt * 16 + ln;
        const float dp = decp[(b << 9) + n];
        const float wc = wcs[n];
        const float bi = bias[n];
        const float vv = vw[n];

        // one-fragment-deep software pipeline over the 16 K-tiles
        v8f acc = {};
        uint4 c0 = *(const uint4*)(sbuf + (lane << 3));
        uint4 c1 = *(const uint4*)(sbuf + (lane << 3) + 4);
#pragma unroll
        for (int kt = 0; kt < 16; ++kt) {
            uint4 n0, n1;
            if (kt + 1 < 16) {
                const uint32_t* bp = sbuf + (((kt + 1) * 32 + lane) << 3);
                n0 = *(const uint4*)(bp);
                n1 = *(const uint4*)(bp + 4);
            }
            union { uint32_t u[8]; v16bf v; } bbu;
            bbu.u[0] = c0.x; bbu.u[1] = c0.y; bbu.u[2] = c0.z; bbu.u[3] = c0.w;
            bbu.u[4] = c1.x; bbu.u[5] = c1.y; bbu.u[6] = c1.z; bbu.u[7] = c1.w;
            acc = __builtin_amdgcn_wmma_f32_16x16x32_bf16(
                false, a[kt], false, bbu.v, (short)0, acc, false, false);
            if (kt + 1 < 16) { c0 = n0; c1 = n1; }
        }

        // fused epilogue: x = tanh(proj + dec + cov*wcs + b); att += x*v_w
#pragma unroll
        for (int j = 0; j < 8; ++j) {
            float x = tanhf(acc[j] + dp + covm[j] * wc + bi);
            att[j] += x * vv;
        }
    }

    // reduce over the 16 lanes of each half-wave (same m, different n)
#pragma unroll
    for (int j = 0; j < 8; ++j) {
        float v = att[j];
        v += __shfl_xor(v, 1);
        v += __shfl_xor(v, 2);
        v += __shfl_xor(v, 4);
        v += __shfl_xor(v, 8);
        if (ln == 0)
            logits[row0 + j + 8 * half] = v + vb[0];
    }
}

// ---------------------------------------------------------------------------
// Masked softmax over S per batch row + coverage update.
// ---------------------------------------------------------------------------
__global__ void softmax_cov(const float* __restrict__ logits,
                            const int*   __restrict__ lens,
                            const float* __restrict__ cov,
                            float*       __restrict__ out) {
    const int b   = blockIdx.x;
    const int tid = threadIdx.x;
    __shared__ float red[256];
    const int len = lens[b];

    float mx = -__builtin_inff();
    for (int s = tid; s < Sdim; s += 256)
        if (s < len) mx = fmaxf(mx, logits[b * Sdim + s]);
    red[tid] = mx; __syncthreads();
    for (int off = 128; off; off >>= 1) {
        if (tid < off) red[tid] = fmaxf(red[tid], red[tid + off]);
        __syncthreads();
    }
    mx = red[0]; __syncthreads();

    float sm = 0.f;
    for (int s = tid; s < Sdim; s += 256)
        if (s < len) sm += __expf(logits[b * Sdim + s] - mx);
    red[tid] = sm; __syncthreads();
    for (int off = 128; off; off >>= 1) {
        if (tid < off) red[tid] += red[tid + off];
        __syncthreads();
    }
    sm = red[0];
    const float inv = 1.0f / sm;

    for (int s = tid; s < Sdim; s += 256) {
        float aw = (s < len) ? __expf(logits[b * Sdim + s] - mx) * inv : 0.0f;
        out[b * Sdim + s] = aw;
        out[Bsz * Sdim + b * Sdim + s] = cov[b * Sdim + s] + aw;
    }
}

// ---------------------------------------------------------------------------
extern "C" void kernel_launch(void* const* d_in, const int* in_sizes, int n_in,
                              void* d_out, int out_size, void* d_ws, size_t ws_size,
                              hipStream_t stream) {
    const float* dec_input    = (const float*)d_in[0];   // 32*1*512
    const float* enc_output   = (const float*)d_in[1];   // 32*2048*512
    const int*   text_lengths = (const int*)  d_in[2];   // 32
    const float* coverage     = (const float*)d_in[3];   // 32*2048
    const float* W            = (const float*)d_in[4];   // 1536*512
    const float* bias         = (const float*)d_in[5];   // 512
    const float* v_w          = (const float*)d_in[6];   // 512
    const float* v_b          = (const float*)d_in[7];   // 1
    float* out = (float*)d_out;                          // 2 * 32*2048

    // workspace layout (dwords): WB[131072] | wcs[512] | decp[16384] | logits[65536]
    uint32_t* WB     = (uint32_t*)d_ws;
    float*    wcs    = (float*)d_ws + 131072;
    float*    decp   = wcs + 512;
    float*    logits = decp + Bsz * Hdim;

    prep_wb  <<<512, 256, 0, stream>>>(W, WB);
    prep_wcs <<<2,   256, 0, stream>>>(W, wcs);
    prep_decp<<<64,  256, 0, stream>>>(W, dec_input, decp);
    attn_main<<<512, 256, 0, stream>>>(enc_output, WB, wcs, decp, coverage,
                                       bias, v_w, v_b, logits);
    softmax_cov<<<Bsz, 256, 0, stream>>>(logits, text_lengths, coverage, out);
}
